// Rainfusion_blockwise_58394375356732
// MI455X (gfx1250) — compile-verified, matrix operand order checked by reading, not run on hardware
//
#include <hip/hip_runtime.h>
#include <stdint.h>

// ---------------------------------------------------------------------------
// Rainfusion block-sparse attention for MI455X (gfx1250, wave32, WMMA).
// SEQ=4608 tokens, 8 heads, d=128, POOL=128 -> 36 blocks, keep=4, prot=6.
// bf16 WMMA (v_wmma_f32_16x16x32_bf16) + f32 softmax/accum.
// K tiles double-buffered via GLOBAL_LOAD_ASYNC_TO_LDS_B128 (ASYNCcnt),
// next+1 tiles prefetched into L2 (global_prefetch_b8).
// ---------------------------------------------------------------------------

#define SEQLEN 4608
#define NHEAD  8
#define HD     128
#define NBLK   36     // SEQ / 128
#define FFL    384
#define TXT    384
#define IMG    3840   // SEQ - FFL - TXT
#define PROT   6      // protected trailing blocks
#define LDSP   136    // 128 + 8 bf16 pad (keeps 16B alignment of rows)

typedef __attribute__((ext_vector_type(16))) __bf16 v16bf;
typedef __attribute__((ext_vector_type(8)))  float  v8f;
typedef __attribute__((ext_vector_type(4)))  unsigned uv4;

union AF { v16bf v; uv4 q[2]; };

__device__ __forceinline__ __bf16 bfbits(unsigned x) {
  unsigned short h = (unsigned short)x;
  __bf16 r; __builtin_memcpy(&r, &h, 2); return r;
}
__device__ __forceinline__ float bf2f(__bf16 b) {
  unsigned short h; __builtin_memcpy(&h, &b, 2);
  unsigned u = ((unsigned)h) << 16;
  float f; __builtin_memcpy(&f, &u, 4); return f;
}
__device__ __forceinline__ __bf16 f2bf(float f) {
  unsigned u; __builtin_memcpy(&u, &f, 4);
  u += 0x7fffu + ((u >> 16) & 1u);            // round-to-nearest-even
  return bfbits(u >> 16);
}

// qkv2 token t -> original token index (same map for gather-in & scatter-out).
// _rearr: original (f,a,h,i,w,j) dims (5,2,2,8,3,8) -> new order (f,h,w,a,i,j)
__device__ __forceinline__ int src_of(int t) {
  if (t < IMG) {
    int f = t / 768;  int r = t - f * 768;
    int h = r / 384;  r -= h * 384;
    int w = r / 128;  r -= w * 128;
    int a = r / 64;   r -= a * 64;
    int i = r / 8;    int j = r - i * 8;
    return FFL + f * 768 + a * 384 + h * 192 + i * 24 + w * 8 + j;
  } else if (t < IMG + FFL) {
    return t - IMG;                 // f_part: original tokens [0,384)
  }
  return t;                         // text tail unchanged
}

// Async-DMA one 128x128 bf16 tile (global, row-major) into an LDS tile.
// 8 x GLOBAL_LOAD_ASYNC_TO_LDS_B128 per thread (GVS mode), ASYNCcnt += 8/wave.
__device__ __forceinline__ void async_tile_to_lds(const __bf16* base,
                                                  __bf16 (*dst)[LDSP], int tid) {
  #pragma unroll
  for (int it = 0; it < 8; ++it) {
    int idx = tid + it * 256;                       // 2048 x 16B
    int row = idx >> 4, c8 = (idx & 15) * 8;
    unsigned lds = (unsigned)(uintptr_t)&dst[row][c8];  // flat addr low 32b = LDS addr
    unsigned go  = (unsigned)(idx * 16);
    asm volatile("global_load_async_to_lds_b128 %0, %1, %2"
                 :: "v"(lds), "v"(go), "s"(base) : "memory");
  }
}

// ---------------------------------------------------------------------------
// 1) permute + f32->bf16 pack into per-head-contiguous [head][token][128]
// ---------------------------------------------------------------------------
__global__ void pack_kernel(const float* __restrict__ q, const float* __restrict__ k,
                            const float* __restrict__ v,
                            __bf16* __restrict__ Q2, __bf16* __restrict__ K2,
                            __bf16* __restrict__ V2) {
  int t = blockIdx.x;           // qkv2 token
  int n = blockIdx.y;           // head
  int d = threadIdx.x;          // dim
  int s = src_of(t);
  int si = (s * NHEAD + n) * HD + d;               // input  (1,S,N,D)
  size_t di = ((size_t)n * SEQLEN + t) * HD + d;   // packed (N,S,D)
  Q2[di] = f2bf(q[si]);
  K2[di] = f2bf(k[si]);
  V2[di] = f2bf(v[si]);
}

// ---------------------------------------------------------------------------
// 2) 128-token average pool (bf16 read, f32 accumulate)
// ---------------------------------------------------------------------------
__global__ void pool_kernel(const __bf16* __restrict__ Q2, const __bf16* __restrict__ K2,
                            float* __restrict__ Qp, float* __restrict__ Kp) {
  int blk = blockIdx.x, n = blockIdx.y, which = blockIdx.z;
  int d = threadIdx.x;
  const __bf16* srcp = which ? K2 : Q2;
  size_t base = ((size_t)n * SEQLEN + (size_t)blk * 128) * HD + d;
  float s = 0.f;
  for (int i = 0; i < 128; ++i) s += bf2f(srcp[base + (size_t)i * HD]);
  float* dstp = which ? Kp : Qp;
  dstp[(n * NBLK + blk) * HD + d] = s * (1.0f / 128.0f);
}

// ---------------------------------------------------------------------------
// 3) 36x36 pooled block scores + top-4 threshold + protections -> kept lists
//    (top-k on softmax(ps) == top-k on raw scores: softmax is monotone/row)
// ---------------------------------------------------------------------------
__global__ void blockmask_kernel(const float* __restrict__ Qp, const float* __restrict__ Kp,
                                 int* __restrict__ cnt, int* __restrict__ list) {
  int n = blockIdx.x;
  int tid = threadIdx.x;
  __shared__ float sc[NBLK][NBLK];
  for (int idx = tid; idx < NBLK * NBLK; idx += blockDim.x) {
    int l = idx / NBLK, s2 = idx - l * NBLK;
    const float* a = Qp + (n * NBLK + l) * HD;
    const float* b = Kp + (n * NBLK + s2) * HD;
    float d = 0.f;
    #pragma unroll 4
    for (int i = 0; i < HD; ++i) d += a[i] * b[i];
    sc[l][s2] = d;
  }
  __syncthreads();
  if (tid < NBLK) {
    int l = tid;
    float t1 = -1e30f, t2 = -1e30f, t3 = -1e30f, t4 = -1e30f;   // keep = 4
    for (int s2 = 0; s2 < NBLK; ++s2) {
      float x = sc[l][s2];
      if (x > t1)      { t4 = t3; t3 = t2; t2 = t1; t1 = x; }
      else if (x > t2) { t4 = t3; t3 = t2; t2 = x; }
      else if (x > t3) { t4 = t3; t3 = x; }
      else if (x > t4) { t4 = x; }
    }
    int c = 0;
    int* row = list + (n * NBLK + l) * NBLK;
    for (int s2 = 0; s2 < NBLK; ++s2) {
      bool keep = (sc[l][s2] >= t4) || (l >= NBLK - PROT) || (s2 >= NBLK - PROT);
      if (keep) row[c++] = s2;
    }
    cnt[n * NBLK + l] = c;
  }
}

// ---------------------------------------------------------------------------
// 4) flash attention over kept 128x128 blocks. 8 wave32 / WG, 16 q-rows / wave.
//    K double-buffered via async DMA; V loaded+transposed through VGPRs.
// ---------------------------------------------------------------------------
__global__ void __launch_bounds__(256)
attn_kernel(const __bf16* __restrict__ Q2, const __bf16* __restrict__ K2,
            const __bf16* __restrict__ V2, const int* __restrict__ cnt,
            const int* __restrict__ list, float* __restrict__ out) {
  __shared__ __attribute__((aligned(16))) __bf16 sK [2][128][LDSP]; // [buf][ktok][d]
  __shared__ __attribute__((aligned(16))) __bf16 sVt[128][LDSP];    // [d][ktok]
  __shared__ __attribute__((aligned(16))) __bf16 sP [128][LDSP];    // Q stage / P tile

  const int qb = blockIdx.x, n = blockIdx.y;
  const int tid  = threadIdx.x;
  const int lane = tid & 31, wave = tid >> 5;
  const int lr = lane & 15, hi = lane >> 4;
  const int m0 = wave * 16;                 // this wave's query-row strip
  const float scale = 0.088388347648318447f; // 128^-0.5

  const int  c   = cnt[n * NBLK + qb];
  const int* lst = list + (n * NBLK + qb) * NBLK;
  const __bf16* headK = K2 + (size_t)n * SEQLEN * HD;
  const __bf16* headV = V2 + (size_t)n * SEQLEN * HD;

  // kick off async DMA of K block 0 as early as possible
  async_tile_to_lds(headK + (size_t)lst[0] * 128 * HD, sK[0], tid);

  // ---- stage Q tile into sP, then grab A-fragments into registers ----
  const uv4* Qg = (const uv4*)(Q2 + ((size_t)n * SEQLEN + (size_t)qb * 128) * HD);
  #pragma unroll
  for (int it = 0; it < 8; ++it) {
    int idx = tid + it * 256;               // 2048 x 16B = 128x128 bf16
    int row = idx >> 4, c8 = (idx & 15) * 8;
    *(uv4*)&sP[row][c8] = Qg[idx];
  }
  __syncthreads();
  AF qf[4];
  #pragma unroll
  for (int kc = 0; kc < 4; ++kc) {          // A[16x32]: lane lr=row, K split per half
    qf[kc].q[0] = *(const uv4*)&sP[m0 + lr][kc * 32 + 8 * hi];
    qf[kc].q[1] = *(const uv4*)&sP[m0 + lr][kc * 32 + 16 + 8 * hi];
  }
  __syncthreads();                          // sP reused for P below

  v8f  o[8];
  float mo[8], ls[8];
  #pragma unroll
  for (int i = 0; i < 8; ++i) { o[i] = (v8f){0,0,0,0,0,0,0,0}; mo[i] = -1e30f; ls[i] = 0.f; }

  for (int ib = 0; ib < c; ++ib) {
    const int kb = lst[ib];
    // async DMA next K block into the other buffer (overlaps everything below)
    if (ib + 1 < c)
      async_tile_to_lds(headK + (size_t)lst[ib + 1] * 128 * HD, sK[(ib + 1) & 1], tid);
    // prefetch the block after next into L2 (one 32KB tile each for K and V)
    if (ib + 2 < c) {
      const char* pk = (const char*)(headK + (size_t)lst[ib + 2] * 128 * HD);
      const char* pv = (const char*)(headV + (size_t)lst[ib + 2] * 128 * HD);
      __builtin_prefetch(pk + tid * 128, 0, 1);
      __builtin_prefetch(pv + tid * 128, 0, 1);
    }

    // ---- V tile: regular load + in-register transpose into sVt ----
    const uv4* Vg = (const uv4*)(headV + (size_t)kb * 128 * HD);
    #pragma unroll
    for (int it = 0; it < 8; ++it) {
      int idx = tid + it * 256;
      int row = idx >> 4, c8 = (idx & 15) * 8;
      uv4 vv = Vg[idx];
      sVt[c8 + 0][row] = bfbits(vv.x & 0xffffu);
      sVt[c8 + 1][row] = bfbits(vv.x >> 16);
      sVt[c8 + 2][row] = bfbits(vv.y & 0xffffu);
      sVt[c8 + 3][row] = bfbits(vv.y >> 16);
      sVt[c8 + 4][row] = bfbits(vv.z & 0xffffu);
      sVt[c8 + 5][row] = bfbits(vv.z >> 16);
      sVt[c8 + 6][row] = bfbits(vv.w & 0xffffu);
      sVt[c8 + 7][row] = bfbits(vv.w >> 16);
    }
    // wait for THIS block's K DMA: completions are in-order, so <=8 outstanding
    // means only the (ib+1) batch may still be in flight.
    if (ib + 1 < c) asm volatile("s_wait_asynccnt 8" ::: "memory");
    else            asm volatile("s_wait_asynccnt 0" ::: "memory");
    __syncthreads();

    const __bf16 (*sKb)[LDSP] = sK[ib & 1];

    // ---- S = Q K^T for this wave's 16x128 strip (8 tiles x 4 k-chunks) ----
    v8f s[8];
    #pragma unroll
    for (int n0 = 0; n0 < 8; ++n0) {
      v8f acc = (v8f){0,0,0,0,0,0,0,0};
      #pragma unroll
      for (int kc = 0; kc < 4; ++kc) {
        AF b;  // B[32xd x 16col]: b.e[j] = K[col=n0*16+lr][d=kc*32+8hi+j]
        b.q[0] = *(const uv4*)&sKb[n0 * 16 + lr][kc * 32 + 8 * hi];
        b.q[1] = *(const uv4*)&sKb[n0 * 16 + lr][kc * 32 + 16 + 8 * hi];
        acc = __builtin_amdgcn_wmma_f32_16x16x32_bf16(false, qf[kc].v, false, b.v,
                                                      (short)0, acc, false, false);
      }
      #pragma unroll
      for (int r = 0; r < 8; ++r) acc[r] *= scale;
      s[n0] = acc;
    }

    // ---- online softmax; row r lives at (M = r + 8*hi), cols striped on lanes
    float rm[8];
    #pragma unroll
    for (int r = 0; r < 8; ++r) {
      float x = s[0][r];
      #pragma unroll
      for (int n0 = 1; n0 < 8; ++n0) x = fmaxf(x, s[n0][r]);
      x = fmaxf(x, __shfl_xor(x, 1, 32));
      x = fmaxf(x, __shfl_xor(x, 2, 32));
      x = fmaxf(x, __shfl_xor(x, 4, 32));
      x = fmaxf(x, __shfl_xor(x, 8, 32));   // stays inside the 16-lane half
      rm[r] = x;
    }
    #pragma unroll
    for (int r = 0; r < 8; ++r) {
      float nm = fmaxf(mo[r], rm[r]);
      float al = __expf(mo[r] - nm);
      mo[r] = nm;
      ls[r] *= al;
      #pragma unroll
      for (int n0 = 0; n0 < 8; ++n0) o[n0][r] *= al;
    }
    // P = exp(S - m), per-lane partial row sums, bf16 store into own LDS strip
    #pragma unroll
    for (int n0 = 0; n0 < 8; ++n0) {
      #pragma unroll
      for (int r = 0; r < 8; ++r) {
        float p = __expf(s[n0][r] - mo[r]);
        ls[r] += p;
        sP[m0 + r + 8 * hi][n0 * 16 + lr] = f2bf(p);
      }
    }
    asm volatile("s_wait_dscnt 0" ::: "memory");  // own-wave LDS RAW fence

    // ---- O += P V  (A from sP strip, B from transposed V) ----
    AF pa[4];
    #pragma unroll
    for (int kc = 0; kc < 4; ++kc) {
      pa[kc].q[0] = *(const uv4*)&sP[m0 + lr][kc * 32 + 8 * hi];
      pa[kc].q[1] = *(const uv4*)&sP[m0 + lr][kc * 32 + 16 + 8 * hi];
    }
    #pragma unroll
    for (int n0 = 0; n0 < 8; ++n0) {
      #pragma unroll
      for (int kc = 0; kc < 4; ++kc) {
        AF b;  // b.e[j] = V[ktok=kc*32+8hi+j][d=n0*16+lr] = sVt[d][ktok]
        b.q[0] = *(const uv4*)&sVt[n0 * 16 + lr][kc * 32 + 8 * hi];
        b.q[1] = *(const uv4*)&sVt[n0 * 16 + lr][kc * 32 + 16 + 8 * hi];
        o[n0] = __builtin_amdgcn_wmma_f32_16x16x32_bf16(false, pa[kc].v, false, b.v,
                                                        (short)0, o[n0], false, false);
      }
    }
    __syncthreads();   // protect sVt / retire buffer before next iteration
  }

  // ---- finalize: reduce row sums across the 16-lane half, scatter output ----
  #pragma unroll
  for (int r = 0; r < 8; ++r) {
    float x = ls[r];
    x += __shfl_xor(x, 1, 32);
    x += __shfl_xor(x, 2, 32);
    x += __shfl_xor(x, 4, 32);
    x += __shfl_xor(x, 8, 32);
    ls[r] = 1.0f / x;
  }
  #pragma unroll
  for (int n0 = 0; n0 < 8; ++n0) {
    #pragma unroll
    for (int r = 0; r < 8; ++r) {
      int row  = qb * 128 + m0 + r + 8 * hi;   // qkv2 token
      int srow = src_of(row);                  // original token (scatter)
      out[((size_t)srow * NHEAD + n) * HD + n0 * 16 + lr] = o[n0][r] * ls[r];
    }
  }
}

// ---------------------------------------------------------------------------
extern "C" void kernel_launch(void* const* d_in, const int* in_sizes, int n_in,
                              void* d_out, int out_size, void* d_ws, size_t ws_size,
                              hipStream_t stream) {
  const float* q = (const float*)d_in[0];
  const float* k = (const float*)d_in[1];
  const float* v = (const float*)d_in[2];
  // d_in[3] (t_b_idx) is unused by the reference computation.
  float* out = (float*)d_out;

  char* ws = (char*)d_ws;
  const size_t tsz = (size_t)NHEAD * SEQLEN * HD * sizeof(__bf16);  // 9.44 MB
  __bf16* Q2 = (__bf16*)(ws);
  __bf16* K2 = (__bf16*)(ws + tsz);
  __bf16* V2 = (__bf16*)(ws + 2 * tsz);
  size_t off = 3 * tsz;
  float* Qp = (float*)(ws + off);            off += (size_t)NHEAD * NBLK * HD * 4;
  float* Kp = (float*)(ws + off);            off += (size_t)NHEAD * NBLK * HD * 4;
  int*   cnt  = (int*)(ws + off);            off += (size_t)NHEAD * NBLK * 4;
  int*   list = (int*)(ws + off);            // NHEAD*NBLK*NBLK ints

  pack_kernel<<<dim3(SEQLEN, NHEAD), 128, 0, stream>>>(q, k, v, Q2, K2, V2);
  pool_kernel<<<dim3(NBLK, NHEAD, 2), 128, 0, stream>>>(Q2, K2, Qp, Kp);
  blockmask_kernel<<<NHEAD, 256, 0, stream>>>(Qp, Kp, cnt, list);
  attn_kernel<<<dim3(NBLK, NHEAD), 256, 0, stream>>>(Q2, K2, V2, cnt, list, out);
}
